// NNUE_86002425135471
// MI455X (gfx1250) — matrix operand order repaired and checked by reading, not compiled, not run on hardware
//
#include <hip/hip_runtime.h>

// ---------------------------------------------------------------------------
// NNUE forward for MI455X (gfx1250, wave32).
// Dominant cost: 2 x [2048,40960]x[40960,256] GEMM = 86 GFLOP over ~756 MB of
// HBM traffic (AI ~114 FLOP/B; HBM floor ~32us @ 23.3 TB/s) -> bf16 WMMA keeps
// it memory-bound. fp32 is streamed once, converted to bf16 while staging into
// double-buffered LDS, computed with v_wmma_f32_16x16x32_bf16 (fp32 accum),
// bias + clipped-ReLU fused in the epilogue.
// Wave tile 32x32 (4 accumulators) -> 16 ds_load_b128 per 8 WMMAs per K-stage,
// one barrier per 64-wide K stage, next stage prefetched into registers.
// ---------------------------------------------------------------------------

typedef __attribute__((ext_vector_type(16))) __bf16 bf16x16;  // 8 VGPRs / lane
typedef __attribute__((ext_vector_type(8)))  float  f32x8;    // WMMA C/D
typedef __attribute__((ext_vector_type(4)))  float  f32x4;
typedef __attribute__((ext_vector_type(4)))  unsigned int u32x4;

union FragU { u32x4 q[2]; bf16x16 v; };   // two 16B LDS loads -> one fragment

#define IN_K   40960
#define H1_N   256
#define COMB_N 512
#define BATCH  2048

#define FT_BM 64
#define FT_BN 64
#define FT_BK 64
#define LDSS  72   // row stride (elements): 144 B = 9*16 B, keeps frags aligned
#define NSTAGES (IN_K / FT_BK)   // 640

__global__ __launch_bounds__(128)
void nnue_ft_wmma(const float* __restrict__ inP, const float* __restrict__ inO,
                  const float* __restrict__ W1p, const float* __restrict__ b1p,
                  const float* __restrict__ W1o, const float* __restrict__ b1o,
                  float* __restrict__ comb)
{
    const int persp = blockIdx.z;
    const float* __restrict__ A    = persp ? inO : inP;   // [2048, 40960]
    const float* __restrict__ W    = persp ? W1o : W1p;   // [256,  40960]
    const float* __restrict__ bias = persp ? b1o : b1p;   // [256]

    const int n0 = blockIdx.x * FT_BN;   // output col within 256
    const int m0 = blockIdx.y * FT_BM;   // batch row

    // Double-buffered staging tiles (ping-pong): 36 KB total.
    __shared__ __align__(16) __bf16 As[2][FT_BM][LDSS];  // m-major [m][k]
    __shared__ __align__(16) __bf16 Bs[2][FT_BN][LDSS];  // n-major [n][k]

    const int tid  = threadIdx.x;        // 128 threads = 4 waves
    const int lane = tid & 31;
    const int wid  = tid >> 5;
    const int wm   = wid & 1;            // 2 wave-rows of 32
    const int wn   = wid >> 1;           // 2 wave-cols of 32
    const int half = lane >> 4;          // lane-group (ISA A/B layouts)
    const int l16  = lane & 15;

    f32x8 acc[2][2] = {};                // 32x32 wave tile: [sm][sn]

    // Staging map: 128 threads, 64 rows x 64 k each for A and B.
    const int ldr = tid >> 1;            // 0..63
    const int ldc = (tid & 1) * 32;      // 0 or 32
    const float* Aptr = A + (size_t)(m0 + ldr) * IN_K + ldc;
    const float* Wptr = W + (size_t)(n0 + ldr) * IN_K + ldc;

    f32x4 ra[8], rw[8];                  // in-flight stage (regs)

    auto stage_load = [&](int k0) {
#pragma unroll
        for (int i = 0; i < 8; ++i) {
            ra[i] = *(const f32x4*)(Aptr + k0 + 4 * i);
            rw[i] = *(const f32x4*)(Wptr + k0 + 4 * i);
        }
    };
    auto stage_store = [&](int buf) {
#pragma unroll
        for (int i = 0; i < 8; ++i) {
            const int c = ldc + 4 * i;
            As[buf][ldr][c + 0] = (__bf16)ra[i].x;
            As[buf][ldr][c + 1] = (__bf16)ra[i].y;
            As[buf][ldr][c + 2] = (__bf16)ra[i].z;
            As[buf][ldr][c + 3] = (__bf16)ra[i].w;
            Bs[buf][ldr][c + 0] = (__bf16)rw[i].x;
            Bs[buf][ldr][c + 1] = (__bf16)rw[i].y;
            Bs[buf][ldr][c + 2] = (__bf16)rw[i].z;
            Bs[buf][ldr][c + 3] = (__bf16)rw[i].w;
        }
    };

    // Prologue: stage 0 into buffer 0.
    stage_load(0);
    stage_store(0);
    __syncthreads();

    for (int it = 0; it < NSTAGES; ++it) {
        const int cur = it & 1;
        if (it + 1 < NSTAGES) stage_load((it + 1) * FT_BK);  // overlap compute

#pragma unroll
        for (int ks = 0; ks < 2; ++ks) {            // two K=32 steps per stage
            // A fragments (16x32 bf16): lane holds row m=l%16;
            // elems 0..7  -> K = ks*32 + half*8 + [0..8)
            // elems 8..15 -> K = ks*32 + 16 + half*8 + [0..8)
            FragU fa[2];
#pragma unroll
            for (int sm = 0; sm < 2; ++sm) {
                const __bf16* ar =
                    &As[cur][wm * 32 + sm * 16 + l16][ks * 32 + half * 8];
                fa[sm].q[0] = *(const u32x4*)(ar);
                fa[sm].q[1] = *(const u32x4*)(ar + 16);
            }
#pragma unroll
            for (int sn = 0; sn < 2; ++sn) {
                // B fragment (32x16 bf16): lane holds col n=l%16;
                // elems 0..7 -> K = ks*32 + half*8 + [0..8); elems 8..15 -> K = ks*32 + 16 + half*8 + [0..8)
                FragU fb;
                const __bf16* br =
                    &Bs[cur][wn * 32 + sn * 16 + l16][ks * 32 + half * 8];
                fb.q[0] = *(const u32x4*)(br);
                fb.q[1] = *(const u32x4*)(br + 16);
#pragma unroll
                for (int sm = 0; sm < 2; ++sm) {
                    acc[sm][sn] = __builtin_amdgcn_wmma_f32_16x16x32_bf16(
                        /*neg_a=*/false, fa[sm].v, /*neg_b=*/false, fb.v,
                        /*c_mod=*/(short)0, acc[sm][sn],
                        /*reuse_a=*/false, /*reuse_b=*/false);
                }
            }
        }

        if (it + 1 < NSTAGES) stage_store((it + 1) & 1);  // other buffer
        __syncthreads();   // one barrier per stage (double buffering)
    }

    // Epilogue: C/D layout (ISA 7.12.2): VGPR r, lane l -> M=(l/16)*8+r, N=l%16.
#pragma unroll
    for (int sm = 0; sm < 2; ++sm) {
        const int crow0 = m0 + wm * 32 + sm * 16 + half * 8;
#pragma unroll
        for (int sn = 0; sn < 2; ++sn) {
            const f32x8 c = acc[sm][sn];
            const int col_local = n0 + wn * 32 + sn * 16 + l16;  // within 256
            const float bv = bias[col_local];
            const int gcol = persp * H1_N + col_local;           // within 512
#pragma unroll
            for (int r = 0; r < 8; ++r) {
                float v = c[r] + bv;
                v = fminf(fmaxf(v, 0.0f), 1.0f);                 // clipped ReLU
                comb[(size_t)(crow0 + r) * COMB_N + gcol] = v;
            }
        }
    }
}

// ---------------------------------------------------------------------------
// Tail MLP: [2048,512] -> 32 -> 32 -> 1 with clipped ReLU. One thread per row;
// weight addresses are lane-uniform -> scalar loads, broadcast to the wave.
// (~34 M MAC total: negligible next to the 43 G MAC feature transformer.)
// ---------------------------------------------------------------------------
__global__ __launch_bounds__(256)
void nnue_tail(const float* __restrict__ comb,
               const float* __restrict__ W2, const float* __restrict__ b2,
               const float* __restrict__ W3, const float* __restrict__ b3,
               const float* __restrict__ Wo, const float* __restrict__ bo,
               float* __restrict__ out)
{
    const int row = blockIdx.x * blockDim.x + threadIdx.x;
    const float* crow = comb + (size_t)row * COMB_N;

    float h1[32];
#pragma unroll
    for (int j = 0; j < 32; ++j) h1[j] = b2[j];
    for (int k = 0; k < COMB_N; ++k) {
        const float x = crow[k];
#pragma unroll
        for (int j = 0; j < 32; ++j) h1[j] = fmaf(x, W2[j * COMB_N + k], h1[j]);
    }
#pragma unroll
    for (int j = 0; j < 32; ++j) h1[j] = fminf(fmaxf(h1[j], 0.0f), 1.0f);

    float h2[32];
#pragma unroll
    for (int j = 0; j < 32; ++j) h2[j] = b3[j];
#pragma unroll
    for (int k = 0; k < 32; ++k) {
        const float x = h1[k];
#pragma unroll
        for (int j = 0; j < 32; ++j) h2[j] = fmaf(x, W3[j * 32 + k], h2[j]);
    }

    float o = bo[0];
#pragma unroll
    for (int j = 0; j < 32; ++j)
        o = fmaf(fminf(fmaxf(h2[j], 0.0f), 1.0f), Wo[j], o);
    out[row] = fminf(fmaxf(o, 0.0f), 1.0f);
}

extern "C" void kernel_launch(void* const* d_in, const int* in_sizes, int n_in,
                              void* d_out, int out_size, void* d_ws, size_t ws_size,
                              hipStream_t stream)
{
    const float* inP = (const float*)d_in[0];
    const float* inO = (const float*)d_in[1];
    const float* W1p = (const float*)d_in[2];
    const float* b1p = (const float*)d_in[3];
    const float* W1o = (const float*)d_in[4];
    const float* b1o = (const float*)d_in[5];
    const float* W2  = (const float*)d_in[6];
    const float* b2  = (const float*)d_in[7];
    const float* W3  = (const float*)d_in[8];
    const float* b3  = (const float*)d_in[9];
    const float* Wo  = (const float*)d_in[10];
    const float* bo  = (const float*)d_in[11];

    float* comb = (float*)d_ws;   // [2048, 512] fp32 combined accumulator

    dim3 g1(H1_N / FT_BN, BATCH / FT_BM, 2);   // (4, 32, 2) = 256 workgroups
    nnue_ft_wmma<<<g1, 128, 0, stream>>>(inP, inO, W1p, b1p, W1o, b1o, comb);

    nnue_tail<<<BATCH / 256, 256, 0, stream>>>(comb, W2, b2, W3, b3, Wo, bo,
                                               (float*)d_out);
}
